// MortonWindowTransformer_22866405884041
// MI455X (gfx1250) — matrix operand order repaired and checked by reading, not compiled, not run on hardware
//
#include <hip/hip_runtime.h>
#include <hip/hip_bf16.h>
#include <math.h>

// ---------------- problem constants (match reference) ----------------
#define NPTS 65536
#define DMODEL 256
#define NHEAD 8
#define DHEAD 32            // DMODEL / NHEAD
#define NLAYER 4
#define FFDIM 2048
#define WSZ 128             // window size
#define NWIN (NPTS / WSZ)   // 512 windows
#define MBITS 7
#define LN_EPS 1e-5f

typedef __attribute__((ext_vector_type(16))) __bf16        v16bf;
typedef __attribute__((ext_vector_type(8)))  float         v8f;
typedef __attribute__((ext_vector_type(4)))  unsigned int  v4u;
typedef __attribute__((ext_vector_type(8)))  int           v8i;
typedef __attribute__((ext_vector_type(4)))  int           v4i;

// =====================================================================
// TDM helper: issue a 2D tile load (rows x cols fp32, row-major with
// row stride `stride0` elements) from global memory into LDS.
// Descriptor layout per CDNA5 ISA:
//   group0: [1:0]=count(1), [63:32]=lds_addr, [120:64]=global_addr,
//           [127:126]=type(2)
//   group1: [17:16]=data_size(2 -> 4B), [79:48]=tensor_dim0,
//           [111:80]=tensor_dim1, [127:112]=tile_dim0,
//           [143:128]=tile_dim1, [207:160]=tensor_dim0_stride
// Issued from a single wave (EXEC is ignored; one DMA per executing wave).
// This toolchain exposes the 6-arg builtin (extra int32x8 group + cpol).
// =====================================================================
__device__ __forceinline__
void tdm_load_2d(unsigned lds_addr, const void* gptr,
                 unsigned tile_d0, unsigned tile_d1,
                 unsigned long long stride0,
                 unsigned tensor_d0, unsigned tensor_d1) {
    unsigned long long ga = (unsigned long long)(uintptr_t)gptr;
    v4u g0;
    g0[0] = 1u;                                         // count=1 (valid)
    g0[1] = lds_addr;                                   // LDS byte address
    g0[2] = (unsigned)(ga & 0xffffffffu);               // global_addr[31:0]
    g0[3] = (unsigned)((ga >> 32) & 0x01ffffffu)        // global_addr[56:32]
          | (2u << 30);                                 // type = 2 ("image")
    v8i g1;
    g1[0] = (int)(2u << 16);                            // data_size=2 -> 4 bytes
    g1[1] = (int)((tensor_d0 & 0xffffu) << 16);         // tensor_dim0[15:0]
    g1[2] = (int)((tensor_d0 >> 16) |
                  ((tensor_d1 & 0xffffu) << 16));       // dim0[31:16]|dim1[15:0]
    g1[3] = (int)((tensor_d1 >> 16) |
                  ((tile_d0 & 0xffffu) << 16));         // dim1[31:16]|tile_dim0
    g1[4] = (int)(tile_d1 & 0xffffu);                   // tile_dim1 (tile_dim2=0)
    g1[5] = (int)(stride0 & 0xffffffffu);               // tensor_dim0_stride lo
    g1[6] = (int)((stride0 >> 32) & 0xffffu);           // stride hi (dim1_stride=0)
    g1[7] = 0;
    v4i z4 = {0, 0, 0, 0};
    v8i z8 = {0, 0, 0, 0, 0, 0, 0, 0};
    __builtin_amdgcn_tensor_load_to_lds(g0, g1, z4, z4, z8, 0);
}

// =====================================================================
// 1) Morton keys:  key = batch * 2^21  |  interleave(x,y,z) (21 bits)
// =====================================================================
__global__ void morton_kernel(const int* __restrict__ coords,
                              int* __restrict__ keys, int n) {
    int i = blockIdx.x * 256 + threadIdx.x;
    if (i >= n) return;
    int b = coords[i * 4 + 0];
    int x = coords[i * 4 + 1];
    int y = coords[i * 4 + 2];
    int z = coords[i * 4 + 3];
    int m = 0;
#pragma unroll
    for (int t = 0; t < MBITS; ++t) {
        m |= ((x >> t) & 1) << (3 * t + 2);
        m |= ((y >> t) & 1) << (3 * t + 1);
        m |= ((z >> t) & 1) << (3 * t + 0);
    }
    keys[i] = b * (1 << (3 * MBITS)) + m;
}

// =====================================================================
// 2) Stable rank sort (matches jnp.argsort stability):
//    rank(i) = #{ j : key[j] < key[i]  or (key[j]==key[i] and j<i) }
//    idx[rank(i)] = i
// =====================================================================
__global__ __launch_bounds__(256)
void rank_sort_kernel(const int* __restrict__ keys,
                      int* __restrict__ idx, int n) {
    __shared__ int sk[256];
    int i  = blockIdx.x * 256 + threadIdx.x;
    int my = keys[i];
    int rank = 0;
    for (int base = 0; base < n; base += 256) {
        __syncthreads();
        sk[threadIdx.x] = keys[base + threadIdx.x];
        __syncthreads();
#pragma unroll 8
        for (int j = 0; j < 256; ++j) {
            int kj = sk[j];
            rank += (kj < my) | ((kj == my) & ((base + j) < i));
        }
    }
    idx[rank] = i;
}

// =====================================================================
// 3) Gather sorted features + positional embeddings
// =====================================================================
__global__ __launch_bounds__(256)
void gather_pe_kernel(const int* __restrict__ idx,
                      const int* __restrict__ coords,
                      const float* __restrict__ feat,
                      const float* __restrict__ pex,
                      const float* __restrict__ pey,
                      const float* __restrict__ pez,
                      const float* __restrict__ pes,
                      float* __restrict__ x) {
    int pos = blockIdx.x;
    int d   = threadIdx.x;
    int src = idx[pos];
    int cx = coords[src * 4 + 1];
    int cy = coords[src * 4 + 2];
    int cz = coords[src * 4 + 3];
    x[(size_t)pos * DMODEL + d] =
        feat[(size_t)src * DMODEL + d] +
        pex[(size_t)cx * DMODEL + d] +
        pey[(size_t)cy * DMODEL + d] +
        pez[(size_t)cz * DMODEL + d] +
        pes[(size_t)1 * DMODEL + d];
}

// =====================================================================
// 4) TDM-staged bf16 WMMA GEMM:  C[M,N] = A[M,K] @ W[N,K]^T + bias
//    optional residual add and ReLU.
//    Block: 256 threads = 8 waves, block tile 64(M) x 128(N),
//    wave tile 16(M) x 64(N) = 4 accumulator frags, K-step 32.
//    Per K-step, wave 0 issues two Tensor-Data-Mover 2D tile loads
//    (A: 64x32 fp32, B: 128x32 fp32) into LDS, waits TENSORcnt,
//    then all waves build bf16 fragments from LDS (ISA 7.12.2 layout):
//      A(16x32 bf16): row = lane%16 ; K = (e/8)*16 + (lane/16)*8 + e%8
//      B(32x16 bf16): col = lane%16 ; K = (e/8)*16 + (lane/16)*8 + e%8
//      C(16x16 f32):  row = r + 8*(lane/16) ; col = lane%16
// =====================================================================
__global__ __launch_bounds__(256)
void gemm_bf16_kernel(const float* __restrict__ A,
                      const float* __restrict__ W,
                      const float* __restrict__ bias,
                      const float* __restrict__ res,
                      float* __restrict__ C,
                      int M, int Nn, int K, int relu) {
    __shared__ float astage[64][32];    //  8 KB: A tile for this K-step
    __shared__ float bstage[128][32];   // 16 KB: W tile for this K-step

    const int wave = threadIdx.x >> 5;
    const int lane = threadIdx.x & 31;
    const int lh   = lane >> 4;   // lane half
    const int ll   = lane & 15;
    const int mstrip = wave & 3;
    const int ngroup = wave >> 2;
    const int Mblk  = blockIdx.x * 64;
    const int Nblk  = blockIdx.y * 128;

    v8f c0 = {}, c1 = {}, c2 = {}, c3 = {};

    const int arow_l = mstrip * 16 + ll;            // A row within block tile

    for (int k0 = 0; k0 < K; k0 += 32) {
        __syncthreads();                            // LDS free from prev step
        if (threadIdx.x < 32) {                     // wave 0 drives the TDM
            tdm_load_2d((unsigned)(uintptr_t)&astage[0][0],
                        A + (size_t)Mblk * K + k0,
                        32, 64, (unsigned long long)K,
                        (unsigned)(K - k0), 64u);
            tdm_load_2d((unsigned)(uintptr_t)&bstage[0][0],
                        W + (size_t)Nblk * K + k0,
                        32, 128, (unsigned long long)K,
                        (unsigned)(K - k0), 128u);
            __builtin_amdgcn_s_wait_tensorcnt(0);
        }
        __syncthreads();                            // tiles visible to all

        // ---- A fragment (shared by the 4 N-subtiles) ----
        v16bf a;
#pragma unroll
        for (int e = 0; e < 8; ++e) a[e]     = (__bf16)astage[arow_l][lh * 8 + e];
#pragma unroll
        for (int e = 0; e < 8; ++e) a[8 + e] = (__bf16)astage[arow_l][16 + lh * 8 + e];

        // ---- 4 B fragments + WMMAs ----
#pragma unroll
        for (int sub = 0; sub < 4; ++sub) {
            const int brow = ngroup * 64 + sub * 16 + ll;
            v16bf b;
#pragma unroll
            for (int e = 0; e < 16; ++e) b[e] = (__bf16)bstage[brow][(e / 8) * 16 + lh * 8 + e % 8];
            v8f* cc = (sub == 0) ? &c0 : (sub == 1) ? &c1 : (sub == 2) ? &c2 : &c3;
            *cc = __builtin_amdgcn_wmma_f32_16x16x32_bf16(
                      false, a, false, b, (short)0, *cc, false, false);
        }
    }

    // ---- epilogue: bias (+residual) (+relu) ----
    v8f acc[4] = { c0, c1, c2, c3 };
#pragma unroll
    for (int sub = 0; sub < 4; ++sub) {
        int n  = Nblk + ngroup * 64 + sub * 16 + ll;
        float bv = bias[n];
#pragma unroll
        for (int r = 0; r < 8; ++r) {
            int m = Mblk + mstrip * 16 + r + 8 * lh;
            float v = acc[sub][r] + bv;
            if (res)  v += res[(size_t)m * Nn + n];
            if (relu) v  = fmaxf(v, 0.0f);
            C[(size_t)m * Nn + n] = v;
        }
    }
}

// =====================================================================
// 5) Windowed attention: one block per (window, head).
//    Each of 8 waves owns a 16-row query strip:
//      scores(16x128) = Q(16x32) @ K^T  -> 8 WMMAs
//      softmax in fp32 (lane-half shuffle reductions)
//      stage probs as bf16 in LDS, reload as A-fragments
//      out(16x32) = P(16x128) @ V(128x32) -> 4x2 WMMAs
// =====================================================================
__global__ __launch_bounds__(256)
void attn_kernel(const float* __restrict__ qkv,
                 float* __restrict__ out) {
    __shared__ __bf16 satt[8][16][WSZ];   // 32 KB

    const int w    = blockIdx.x / NHEAD;
    const int h    = blockIdx.x % NHEAD;
    const int wave = threadIdx.x >> 5;
    const int lane = threadIdx.x & 31;
    const int lh   = lane >> 4;
    const int ll   = lane & 15;
    const int qbase = wave * 16;
    const float* base = qkv + (size_t)w * WSZ * (3 * DMODEL);

    // ---- Q fragment for this wave's strip ----
    v16bf aq;
    {
        const float* qrow = base + (size_t)(qbase + ll) * (3 * DMODEL) + h * DHEAD;
#pragma unroll
        for (int e = 0; e < 8; ++e) aq[e]     = (__bf16)qrow[lh * 8 + e];
#pragma unroll
        for (int e = 0; e < 8; ++e) aq[8 + e] = (__bf16)qrow[16 + lh * 8 + e];
    }

    // ---- scores = Q @ K^T over 8 key tiles ----
    v8f s[8];
#pragma unroll
    for (int j = 0; j < 8; ++j) {
        const float* kr = base + (size_t)(j * 16 + ll) * (3 * DMODEL)
                        + DMODEL + h * DHEAD + lh * 8;
        v16bf bk;
#pragma unroll
        for (int e = 0; e < 16; ++e) bk[e] = (__bf16)kr[(e / 8) * 16 + e % 8];
        v8f z = {};
        s[j] = __builtin_amdgcn_wmma_f32_16x16x32_bf16(
                   false, aq, false, bk, (short)0, z, false, false);
    }

    // ---- softmax over each of the 8 rows this lane-half holds ----
    const float scale = 0.17677669529663687f;   // 1/sqrt(32)
#pragma unroll
    for (int r = 0; r < 8; ++r) {
        float mx = -3.0e38f;
#pragma unroll
        for (int j = 0; j < 8; ++j) mx = fmaxf(mx, s[j][r]);
        for (int off = 1; off < 16; off <<= 1)
            mx = fmaxf(mx, __shfl_xor(mx, off, 16));
        float sum = 0.0f;
        float p[8];
#pragma unroll
        for (int j = 0; j < 8; ++j) {
            p[j] = __expf(scale * (s[j][r] - mx));
            sum += p[j];
        }
        for (int off = 1; off < 16; off <<= 1)
            sum += __shfl_xor(sum, off, 16);
        float inv = 1.0f / sum;
        int row = r + 8 * lh;                    // row within strip
#pragma unroll
        for (int j = 0; j < 8; ++j)
            satt[wave][row][j * 16 + ll] = (__bf16)(p[j] * inv);
    }
    __syncthreads();

    // ---- out = P @ V ----
    v8f o0 = {}, o1 = {};
#pragma unroll
    for (int kt = 0; kt < 4; ++kt) {            // K = 128 in steps of 32
        v16bf aa;
#pragma unroll
        for (int e = 0; e < 8; ++e)
            aa[e]     = satt[wave][ll][kt * 32 + lh * 8 + e];
#pragma unroll
        for (int e = 0; e < 8; ++e)
            aa[8 + e] = satt[wave][ll][kt * 32 + 16 + lh * 8 + e];
#pragma unroll
        for (int nt = 0; nt < 2; ++nt) {
            const int kb = kt * 32 + lh * 8;
            v16bf bv;
#pragma unroll
            for (int e = 0; e < 16; ++e)
                bv[e] = (__bf16)base[(size_t)(kb + (e / 8) * 16 + e % 8) * (3 * DMODEL)
                                     + 2 * DMODEL + h * DHEAD + nt * 16 + ll];
            v8f* oo = nt ? &o1 : &o0;
            *oo = __builtin_amdgcn_wmma_f32_16x16x32_bf16(
                      false, aa, false, bv, (short)0, *oo, false, false);
        }
    }

    // ---- write out strip: out[w*128 + q][h*32 + dh] ----
    v8f oarr[2] = { o0, o1 };
#pragma unroll
    for (int nt = 0; nt < 2; ++nt)
#pragma unroll
        for (int r = 0; r < 8; ++r)
            out[(size_t)(w * WSZ + qbase + r + 8 * lh) * DMODEL
                + h * DHEAD + nt * 16 + ll] = oarr[nt][r];
}

// =====================================================================
// 6) LayerNorm: one wave per row of 256.
// =====================================================================
__global__ __launch_bounds__(256)
void ln_kernel(const float* __restrict__ in,
               const float* __restrict__ g,
               const float* __restrict__ b,
               float* __restrict__ out) {
    int row  = blockIdx.x * 8 + (threadIdx.x >> 5);
    int lane = threadIdx.x & 31;
    const float* r = in + (size_t)row * DMODEL;
    float v[8];
    float sum = 0.0f;
#pragma unroll
    for (int i = 0; i < 8; ++i) { v[i] = r[lane + 32 * i]; sum += v[i]; }
    for (int off = 1; off < 32; off <<= 1) sum += __shfl_xor(sum, off, 32);
    float mu = sum * (1.0f / DMODEL);
    float var = 0.0f;
#pragma unroll
    for (int i = 0; i < 8; ++i) { float d = v[i] - mu; var += d * d; }
    for (int off = 1; off < 32; off <<= 1) var += __shfl_xor(var, off, 32);
    var *= (1.0f / DMODEL);
    float rs = rsqrtf(var + LN_EPS);
#pragma unroll
    for (int i = 0; i < 8; ++i) {
        int col = lane + 32 * i;
        out[(size_t)row * DMODEL + col] = (v[i] - mu) * rs * g[col] + b[col];
    }
}

// =====================================================================
// Host-side orchestration
// =====================================================================
extern "C" void kernel_launch(void* const* d_in, const int* in_sizes, int n_in,
                              void* d_out, int out_size, void* d_ws, size_t ws_size,
                              hipStream_t stream) {
    (void)in_sizes; (void)n_in; (void)out_size; (void)ws_size;

    const int*   coords = (const int*)  d_in[0];
    const float* feat   = (const float*)d_in[1];
    const float* pex    = (const float*)d_in[2];
    const float* pey    = (const float*)d_in[3];
    const float* pez    = (const float*)d_in[4];
    const float* pes    = (const float*)d_in[5];
    const float* Wqkv   = (const float*)d_in[6];
    const float* bqkv   = (const float*)d_in[7];
    const float* Wo     = (const float*)d_in[8];
    const float* bo     = (const float*)d_in[9];
    const float* W1     = (const float*)d_in[10];
    const float* b1     = (const float*)d_in[11];
    const float* W2     = (const float*)d_in[12];
    const float* b2     = (const float*)d_in[13];
    const float* g1     = (const float*)d_in[14];
    const float* be1    = (const float*)d_in[15];
    const float* g2     = (const float*)d_in[16];
    const float* be2    = (const float*)d_in[17];

    // workspace layout
    char* p = (char*)d_ws;
    int*   keys = (int*)p;   p += (size_t)NPTS * 4;
    int*   idx  = (int*)p;   p += (size_t)NPTS * 4;
    float* x    = (float*)p; p += (size_t)NPTS * DMODEL * 4;        // 64 MB
    float* tmp  = (float*)p; p += (size_t)NPTS * DMODEL * 4;        // 64 MB
    float* attn = (float*)p; p += (size_t)NPTS * DMODEL * 4;        // 64 MB
    float* qkv  = (float*)p; p += (size_t)NPTS * 3 * DMODEL * 4;    // 192 MB
    float* ff   = (float*)p;                                        // 512 MB

    // 1) morton keys
    morton_kernel<<<NPTS / 256, 256, 0, stream>>>(coords, keys, NPTS);
    // 2) stable sort -> permutation
    rank_sort_kernel<<<NPTS / 256, 256, 0, stream>>>(keys, idx, NPTS);
    // 3) gather + positional embedding
    gather_pe_kernel<<<NPTS, 256, 0, stream>>>(idx, coords, feat,
                                               pex, pey, pez, pes, x);

    const int M = NPTS;
    for (int l = 0; l < NLAYER; ++l) {
        const float* Wqkv_l = Wqkv + (size_t)l * 3 * DMODEL * DMODEL;
        const float* bqkv_l = bqkv + (size_t)l * 3 * DMODEL;
        const float* Wo_l   = Wo   + (size_t)l * DMODEL * DMODEL;
        const float* bo_l   = bo   + (size_t)l * DMODEL;
        const float* W1_l   = W1   + (size_t)l * FFDIM * DMODEL;
        const float* b1_l   = b1   + (size_t)l * FFDIM;
        const float* W2_l   = W2   + (size_t)l * DMODEL * FFDIM;
        const float* b2_l   = b2   + (size_t)l * DMODEL;
        const float* g1_l   = g1   + (size_t)l * DMODEL;
        const float* be1_l  = be1  + (size_t)l * DMODEL;
        const float* g2_l   = g2   + (size_t)l * DMODEL;
        const float* be2_l  = be2  + (size_t)l * DMODEL;

        // qkv = x @ Wqkv^T + bqkv          (65536 x 768)
        gemm_bf16_kernel<<<dim3(M / 64, (3 * DMODEL) / 128), 256, 0, stream>>>(
            x, Wqkv_l, bqkv_l, nullptr, qkv, M, 3 * DMODEL, DMODEL, 0);

        // windowed multi-head attention
        attn_kernel<<<NWIN * NHEAD, 256, 0, stream>>>(qkv, attn);

        // tmp = x + attn @ Wo^T + bo ; x = LN1(tmp)
        gemm_bf16_kernel<<<dim3(M / 64, DMODEL / 128), 256, 0, stream>>>(
            attn, Wo_l, bo_l, x, tmp, M, DMODEL, DMODEL, 0);
        ln_kernel<<<M / 8, 256, 0, stream>>>(tmp, g1_l, be1_l, x);

        // ff = relu(x @ W1^T + b1)         (65536 x 2048)
        gemm_bf16_kernel<<<dim3(M / 64, FFDIM / 128), 256, 0, stream>>>(
            x, W1_l, b1_l, nullptr, ff, M, FFDIM, DMODEL, 1);

        // tmp = x + ff @ W2^T + b2 ; x (or d_out on last layer) = LN2(tmp)
        gemm_bf16_kernel<<<dim3(M / 64, DMODEL / 128), 256, 0, stream>>>(
            ff, W2_l, b2_l, x, tmp, M, DMODEL, FFDIM, 0);
        float* lnout = (l == NLAYER - 1) ? (float*)d_out : x;
        ln_kernel<<<M / 8, 256, 0, stream>>>(tmp, g2_l, be2_l, lnout);
    }
}
